// MultiHeadSelfAttention_72782515798646
// MI455X (gfx1250) — matrix operand order repaired and checked by reading, not compiled, not run on hardware
//
#include <hip/hip_runtime.h>

typedef __bf16 bf16_t;
typedef __attribute__((ext_vector_type(8)))  __bf16 v8bf;
typedef __attribute__((ext_vector_type(16))) __bf16 v16bf;
typedef __attribute__((ext_vector_type(8)))  float  v8f;
typedef __attribute__((ext_vector_type(4)))  unsigned int v4u;
typedef __attribute__((ext_vector_type(8)))  unsigned int v8u;

#define B_  2
#define T_  1024
#define C_  2048
#define H_  16
#define D_  128
#define P_  1024
#define L_  2048   /* P_ + T_ */

// Build a 16-element A/B fragment from two aligned 16-byte LDS/global chunks.
__device__ __forceinline__ v16bf frag2(const bf16_t* p0, const bf16_t* p1) {
  v8bf lo = *(const v8bf*)p0;
  v8bf hi = *(const v8bf*)p1;
  return __builtin_shufflevector(lo, hi, 0,1,2,3,4,5,6,7,8,9,10,11,12,13,14,15);
}

__device__ __forceinline__ v8f wmma_bf16(v16bf a, v16bf b, v8f c) {
  // (neg_a, A, neg_b, B, c_mod, C, reuse_a, reuse_b)
  return __builtin_amdgcn_wmma_f32_16x16x32_bf16(false, a, false, b, (short)0, c,
                                                 false, false);
}

// ---------------------------------------------------------------------------
// TDM helper: issue a 2D tensor_load_to_lds of a bf16 tile.
//   tile: rows x cols bf16 elements, source row stride = src_stride elems,
//   LDS destination padded +16B after each `pad_iv`-encoded interval.
//   g1 descriptor is built by the caller (loop-invariant); g0 per call.
// Descriptor layout per CDNA5 ISA.
// ---------------------------------------------------------------------------
__device__ __forceinline__ void tdm_load_2d(unsigned lds_addr,
                                            unsigned long long gaddr,
                                            v8u g1) {
  v4u g0;
  g0.x = 1u;                                   // count=1 (valid), user mode
  g0.y = lds_addr;                             // LDS byte address
  g0.z = (unsigned)gaddr;                      // global addr [31:0]
  g0.w = ((unsigned)(gaddr >> 32) & 0x01FFFFFFu) | (2u << 30);  // addr[56:32], type=2
  asm volatile("tensor_load_to_lds %0, %1" :: "s"(g0), "s"(g1) : "memory");
}

__device__ __forceinline__ v8u tdm_group1(unsigned tensor_d0, unsigned tensor_d1,
                                          unsigned tile_d0, unsigned tile_d1,
                                          unsigned stride0, unsigned pad_iv) {
  v8u g1;
  // data_size=1 (2 bytes), pad_enable, pad_interval=pad_iv, pad_amount=3 (16B)
  g1[0] = (1u << 16) | (1u << 20) | (pad_iv << 22) | (3u << 25);
  g1[1] = (tensor_d0 & 0xFFFFu) << 16;                  // tensor_dim0[15:0]
  g1[2] = (tensor_d0 >> 16) | ((tensor_d1 & 0xFFFFu) << 16);
  g1[3] = (tensor_d1 >> 16) | (tile_d0 << 16);          // tile_dim0
  g1[4] = tile_d1;                                      // tile_dim1 (tile_dim2=0)
  g1[5] = stride0;                                      // tensor_dim0_stride[31:0]
  g1[6] = 0u;                                           // stride0 hi / stride1 lo
  g1[7] = 0u;
  return g1;
}

// ---------------------------------------------------------------------------
// GEMM:  Y[M,N] = A[M,K] @ W[N,K]^T   (NT form, both contiguous along K)
// 128x128 workgroup tile, BK=32, 8 waves, each wave computes 16 rows x 128 cols.
// mode 0: outF[m*N+n] = acc + bias[n]                     (final projection)
// mode 1: outBF[b,h,t,d] = bf16(acc*scale)                (Q, scale=1/sqrt(D))
// mode 2: outF / outBF at [b,h,P+t,d]                     (K or V + shadow)
// A is f32 (convert during staging) or bf16 (staged via TDM).
// ---------------------------------------------------------------------------
template <typename AT>
__global__ __launch_bounds__(256)
void gemm_nt(const AT* __restrict__ A, const float* __restrict__ W,
             const float* __restrict__ bias, float* __restrict__ outF,
             bf16_t* __restrict__ outBF, int M, int N, int K,
             int mode, float scale) {
  __shared__ bf16_t Al[128][40];   // [row][k], 80B row stride (16B-aligned)
  __shared__ bf16_t Bl[128][40];   // [n][k]

  const int tid  = threadIdx.x;
  const int lane = tid & 31;
  const int wv   = tid >> 5;           // 0..7
  const int col  = lane & 15;          // B/C column within 16-tile
  const int mb   = (lane >> 4) * 8;    // C-layout row base
  const int klo  = (lane >> 4) * 8;    // A-frag K base
  const int kb16 = (lane >> 4) * 16;   // unused; B-frags also use klo
  const int m0 = blockIdx.y * 128;
  const int n0 = blockIdx.x * 128;

  v8f acc[8];
  #pragma unroll
  for (int ct = 0; ct < 8; ++ct)
    #pragma unroll
    for (int i = 0; i < 8; ++i) acc[ct][i] = 0.f;

  const int arow = tid >> 1;           // 0..127
  const int koff = (tid & 1) * 16;     // 0 or 16

  // TDM descriptor group-1 for bf16 A tiles: 32x128 tile (x-major = K),
  // row stride K, +16B LDS pad per 64B row -> Al[128][40] layout.
  v8u g1a;
  if constexpr (sizeof(AT) == 2)
    g1a = tdm_group1(/*tensor_d0=*/32, /*tensor_d1=*/128,
                     /*tile_d0=*/32, /*tile_d1=*/128,
                     /*stride0=*/(unsigned)K, /*pad_iv=*/3u /*64B*/);

  for (int k0 = 0; k0 < K; k0 += 32) {
    __syncthreads();
    // ---- stage A tile ----
    if constexpr (sizeof(AT) == 2) {
      if (wv == 0) {
        unsigned long long ga = (unsigned long long)(size_t)A +
                                ((unsigned long long)m0 * K + k0) * 2ull;
        tdm_load_2d((unsigned)(size_t)&Al[0][0], ga, g1a);
      }
    } else {
      const AT* ap = A + (size_t)(m0 + arow) * K + k0 + koff;
      const float4* fp = (const float4*)ap;
      float4 f0 = fp[0], f1 = fp[1], f2 = fp[2], f3 = fp[3];
      v8bf a0 = v8bf{(bf16_t)f0.x,(bf16_t)f0.y,(bf16_t)f0.z,(bf16_t)f0.w,
                     (bf16_t)f1.x,(bf16_t)f1.y,(bf16_t)f1.z,(bf16_t)f1.w};
      v8bf a1 = v8bf{(bf16_t)f2.x,(bf16_t)f2.y,(bf16_t)f2.z,(bf16_t)f2.w,
                     (bf16_t)f3.x,(bf16_t)f3.y,(bf16_t)f3.z,(bf16_t)f3.w};
      *(v8bf*)&Al[arow][koff]     = a0;
      *(v8bf*)&Al[arow][koff + 8] = a1;
      __builtin_prefetch(ap + 32, 0, 1);        // global_prefetch_b8 (next BK)
    }
    // ---- stage W tile (n-major so B-frags read contiguously) ----
    {
      const float4* wp =
          (const float4*)(W + (size_t)(n0 + arow) * K + k0 + koff);
      float4 f0 = wp[0], f1 = wp[1], f2 = wp[2], f3 = wp[3];
      v8bf b0 = v8bf{(bf16_t)f0.x,(bf16_t)f0.y,(bf16_t)f0.z,(bf16_t)f0.w,
                     (bf16_t)f1.x,(bf16_t)f1.y,(bf16_t)f1.z,(bf16_t)f1.w};
      v8bf b1 = v8bf{(bf16_t)f2.x,(bf16_t)f2.y,(bf16_t)f2.z,(bf16_t)f2.w,
                     (bf16_t)f3.x,(bf16_t)f3.y,(bf16_t)f3.z,(bf16_t)f3.w};
      *(v8bf*)&Bl[arow][koff]     = b0;
      *(v8bf*)&Bl[arow][koff + 8] = b1;
      __builtin_prefetch((const float*)wp + 32, 0, 1);
    }
    if constexpr (sizeof(AT) == 2) {
      if (wv == 0) __builtin_amdgcn_s_wait_tensorcnt(0);
    }
    __syncthreads();

    // ---- compute: one A-frag, 8 WMMA across the 128-col strip ----
    v16bf a = frag2(&Al[wv*16 + col][klo], &Al[wv*16 + col][16 + klo]);
    #pragma unroll
    for (int ct = 0; ct < 8; ++ct) {
      v16bf b = frag2(&Bl[ct*16 + col][klo], &Bl[ct*16 + col][16 + klo]);
      acc[ct] = wmma_bf16(a, b, acc[ct]);
    }
  }

  // ---- epilogue ----
  #pragma unroll
  for (int ct = 0; ct < 8; ++ct) {
    #pragma unroll
    for (int r = 0; r < 8; ++r) {
      const int m = m0 + wv*16 + mb + r;
      const int n = n0 + ct*16 + col;
      const float v = acc[ct][r];
      if (mode == 0) {
        outF[(size_t)m * N + n] = v + bias[n];
      } else {
        const int bb = m >> 10, t = m & (T_ - 1);
        const int h  = n >> 7,  d = n & (D_ - 1);
        if (mode == 1) {
          size_t qi = ((size_t)(bb*H_ + h) * T_ + t) * D_ + d;
          outBF[qi] = (bf16_t)(v * scale);
        } else {
          size_t ki = ((size_t)(bb*H_ + h) * L_ + (P_ + t)) * D_ + d;
          outF[ki]  = v;
          outBF[ki] = (bf16_t)v;
        }
      }
    }
  }
}

// ---------------------------------------------------------------------------
// Flash attention: 1 workgroup = 4 waves = 64 query rows of one (b,h).
// Per wave: 16 rows, O accumulator 16x128 f32 in VGPRs, online softmax.
// Key loop in tiles of 32: K tile staged by the TDM (tensor_load_to_lds with
// LDS padding matching Kl[32][136]); V^T staged manually (transpose).
// ---------------------------------------------------------------------------
__global__ __launch_bounds__(128)
void flash_attn(const bf16_t* __restrict__ qbf, const bf16_t* __restrict__ kbf,
                const bf16_t* __restrict__ vbf, bf16_t* __restrict__ obf) {
  __shared__ bf16_t Kl[32][136];      // [key][d]; 272B row stride (TDM pad)
  __shared__ bf16_t Vt[128][40];      // [d][key]
  __shared__ bf16_t Pl[4][16][40];    // per-wave P tile [row][key]

  const int tid  = threadIdx.x;
  const int lane = tid & 31;
  const int wv   = tid >> 5;          // 0..3
  const int col  = lane & 15;
  const int mb   = (lane >> 4) * 8;
  const int klo  = (lane >> 4) * 8;
  const int kb16 = (lane >> 4) * 16;
  const int qb = blockIdx.x * 64;
  const int bh = blockIdx.y;          // 0..31
  const int b  = bh >> 4, h = bh & 15;

  const bf16_t* kbase = kbf + (size_t)bh * L_ * D_;
  const bf16_t* vbase = vbf + (size_t)bh * L_ * D_;

  // Q fragments for this wave's 16 rows (4 K-chunks of 32 over D=128)
  const bf16_t* qp =
      qbf + ((size_t)bh * T_ + (qb + wv*16 + col)) * D_;
  v16bf qa[4];
  #pragma unroll
  for (int c = 0; c < 4; ++c)
    qa[c] = frag2(qp + c*32 + klo, qp + c*32 + 16 + klo);

  v8f o[8];
  #pragma unroll
  for (int dt = 0; dt < 8; ++dt)
    #pragma unroll
    for (int i = 0; i < 8; ++i) o[dt][i] = 0.f;
  float mr[8], lr[8];
  #pragma unroll
  for (int r = 0; r < 8; ++r) { mr[r] = -1e30f; lr[r] = 0.f; }

  const int Lmax = P_ + qb + 64;      // causal bound for this query block
  const int skey = tid >> 2;          // V staging: key row 0..31
  const int sd   = (tid & 3) * 32;    // V staging: d base

  // TDM descriptor group-1 for K tiles: 128x32 tile (x-major = d),
  // row stride D, +16B LDS pad per 256B row -> Kl[32][136] layout.
  const v8u g1k = tdm_group1(/*tensor_d0=*/D_, /*tensor_d1=*/32,
                             /*tile_d0=*/D_, /*tile_d1=*/32,
                             /*stride0=*/D_, /*pad_iv=*/5u /*256B*/);

  for (int j0 = 0; j0 < Lmax; j0 += 32) {
    __syncthreads();
    // ---- stage K tile [32][128] via Tensor Data Mover ----
    if (wv == 0) {
      unsigned long long ga = (unsigned long long)(size_t)kbase +
                              (unsigned long long)j0 * (D_ * 2);
      tdm_load_2d((unsigned)(size_t)&Kl[0][0], ga, g1k);
    }
    // ---- stage V tile transposed -> Vt[d][key] ----
    {
      const bf16_t* vp = vbase + (size_t)(j0 + skey) * D_ + sd;
      #pragma unroll
      for (int c = 0; c < 4; ++c) {
        v8bf v = *(const v8bf*)(vp + c*8);
        #pragma unroll
        for (int i = 0; i < 8; ++i) Vt[sd + c*8 + i][skey] = v[i];
      }
      __builtin_prefetch(vp + 32 * D_, 0, 1);
    }
    if (wv == 0) __builtin_amdgcn_s_wait_tensorcnt(0);
    __syncthreads();

    // ---- S = Q K^T : two 16-key tiles, K-loop over D ----
    v8f st[2];
    #pragma unroll
    for (int kt = 0; kt < 2; ++kt) {
      v8f s;
      #pragma unroll
      for (int i = 0; i < 8; ++i) s[i] = 0.f;
      #pragma unroll
      for (int c = 0; c < 4; ++c) {
        v16bf kfr = frag2(&Kl[kt*16 + col][c*32 + klo],
                          &Kl[kt*16 + col][c*32 + 16 + klo]);
        s = wmma_bf16(qa[c], kfr, s);
      }
      st[kt] = s;
    }

    // Only the last couple of tiles can touch the causal diagonal.
    const bool need_mask = (j0 + 31 > P_ + qb);

    // ---- causal mask + online softmax (row reduce across 16 lanes) ----
    #pragma unroll
    for (int r = 0; r < 8; ++r) {
      const int trow  = qb + wv*16 + mb + r;
      float s0 = st[0][r], s1 = st[1][r];
      if (need_mask) {
        const int limit = P_ + trow;
        if (j0 + col > limit)      s0 = -1e30f;
        if (j0 + 16 + col > limit) s1 = -1e30f;
      }
      float mx = fmaxf(s0, s1);
      #pragma unroll
      for (int msk = 8; msk >= 1; msk >>= 1)
        mx = fmaxf(mx, __shfl_xor(mx, msk, 16));
      const float newm = fmaxf(mr[r], mx);
      const float p0 = __expf(s0 - newm);
      const float p1 = __expf(s1 - newm);
      float rs = p0 + p1;
      #pragma unroll
      for (int msk = 8; msk >= 1; msk >>= 1)
        rs += __shfl_xor(rs, msk, 16);
      const float alpha = __expf(mr[r] - newm);
      lr[r] = lr[r] * alpha + rs;
      mr[r] = newm;
      #pragma unroll
      for (int dt = 0; dt < 8; ++dt) o[dt][r] *= alpha;
      Pl[wv][mb + r][col]      = (bf16_t)p0;
      Pl[wv][mb + r][16 + col] = (bf16_t)p1;
    }
    asm volatile("s_wait_dscnt 0" ::: "memory");  // P stores -> P loads (same wave)

    // ---- O += P @ V  (K=32, 8 output d-tiles) ----
    v16bf pa = frag2(&Pl[wv][col][klo], &Pl[wv][col][16 + klo]);
    #pragma unroll
    for (int dt = 0; dt < 8; ++dt) {
      v16bf bv = frag2(&Vt[dt*16 + col][klo], &Vt[dt*16 + col][16 + klo]);
      o[dt] = wmma_bf16(pa, bv, o[dt]);
    }
  }

  // ---- normalize and write O as bf16 in [B,T,C] layout ----
  #pragma unroll
  for (int r = 0; r < 8; ++r) {
    const int trow = qb + wv*16 + mb + r;
    const float inv = 1.0f / lr[r];
    bf16_t* op = obf + ((size_t)(b*T_ + trow)) * C_ + h * D_;
    #pragma unroll
    for (int dt = 0; dt < 8; ++dt)
      op[dt*16 + col] = (bf16_t)(o[dt][r] * inv);
  }
}

// ---------------------------------------------------------------------------
// Copy past K/V into output caches (f32) and bf16 shadows.
// ---------------------------------------------------------------------------
__global__ void copy_past(const float* __restrict__ pk, const float* __restrict__ pv,
                          float* __restrict__ ok, float* __restrict__ ov,
                          bf16_t* __restrict__ kb, bf16_t* __restrict__ vb) {
  size_t idx = (size_t)blockIdx.x * blockDim.x + threadIdx.x;
  const size_t ntot = (size_t)B_ * H_ * P_ * D_;
  if (idx >= ntot) return;
  size_t bh  = idx / ((size_t)P_ * D_);
  size_t rem = idx - bh * (size_t)P_ * D_;
  size_t dst = bh * (size_t)L_ * D_ + rem;
  float k = pk[idx]; ok[dst] = k; kb[dst] = (bf16_t)k;
  float v = pv[idx]; ov[dst] = v; vb[dst] = (bf16_t)v;
}

extern "C" void kernel_launch(void* const* d_in, const int* in_sizes, int n_in,
                              void* d_out, int out_size, void* d_ws, size_t ws_size,
                              hipStream_t stream) {
  (void)in_sizes; (void)n_in; (void)out_size; (void)ws_size;
  const float* x  = (const float*)d_in[0];
  const float* pk = (const float*)d_in[1];
  const float* pv = (const float*)d_in[2];
  const float* Wk = (const float*)d_in[3];
  const float* Wq = (const float*)d_in[4];
  const float* Wv = (const float*)d_in[5];
  const float* Wp = (const float*)d_in[6];
  const float* bp = (const float*)d_in[7];

  float* out  = (float*)d_out;
  float* outk = out  + (size_t)B_ * T_ * C_;
  float* outv = outk + (size_t)B_ * H_ * L_ * D_;

  bf16_t* qbf = (bf16_t*)d_ws;
  bf16_t* kbf = qbf + (size_t)B_ * H_ * T_ * D_;
  bf16_t* vbf = kbf + (size_t)B_ * H_ * L_ * D_;
  bf16_t* obf = vbf + (size_t)B_ * H_ * L_ * D_;   // total ws use: 48 MB

  const int M = B_ * T_, N = C_, K = C_;
  dim3 gg(N / 128, M / 128);                       // 16 x 16 workgroups
  const float qscale = 0.08838834764831845f;       // 1/sqrt(128)

  {
    int n = B_ * H_ * P_ * D_;
    copy_past<<<(n + 255) / 256, 256, 0, stream>>>(pk, pv, outk, outv, kbf, vbf);
  }
  gemm_nt<float><<<gg, 256, 0, stream>>>(x, Wq, nullptr, nullptr, qbf,
                                         M, N, K, 1, qscale);
  gemm_nt<float><<<gg, 256, 0, stream>>>(x, Wk, nullptr, outk, kbf,
                                         M, N, K, 2, 1.f);
  gemm_nt<float><<<gg, 256, 0, stream>>>(x, Wv, nullptr, outv, vbf,
                                         M, N, K, 2, 1.f);
  flash_attn<<<dim3(T_ / 64, B_ * H_), 128, 0, stream>>>(qbf, kbf, vbf, obf);
  gemm_nt<bf16_t><<<gg, 256, 0, stream>>>(obf, Wp, bp, out, nullptr,
                                          M, N, K, 0, 1.f);
}
